// Transformer_23957327577148
// MI455X (gfx1250) — compile-verified, matrix operand order checked by reading, not run on hardware
//
#include <hip/hip_runtime.h>

typedef __attribute__((ext_vector_type(16))) _Float16 v16h;
typedef __attribute__((ext_vector_type(8)))  _Float16 v8h;
typedef __attribute__((ext_vector_type(8)))  float    v8f;

#define H_DIM  512
#define NHEADS 8
#define DPH    64
#define DFF    2048
#define SEQ    512
#define BATCHN 8
#define ROWS   (BATCHN * SEQ)            /* 4096 */
#define VOCABN 32000
#define HTOT   (BATCHN * NHEADS * SEQ * DPH) /* 2097152 */

// ---------------------------------------------------------------------------
// Register-blocked WMMA GEMM:  C = act( scale*(A @ Bt^T) + bias [+ res] )
//   A  : fp32 [M][K] row-major (per-batch stride sA), converted to f16 on load
//   Bt : f16  [N][K] row-major (weights pre-transposed; per-batch stride sB)
//   OUTMODE 0: fp32 [M][N];  1: f16 [M][N];  2: f16 transposed [N][M]
//   ACT 0: none;  1: exact GELU;  2: tanh
// Block = 4 waves stacked along M (block tile 128x64); each wave owns a
// 32x64 sub-tile: 2 A-frags x 4 B-frags -> 8 v_wmma_f32_16x16x32_f16 per
// K-step (~16 FLOP per byte fetched from L2).
// Requires: M % 128 == 0, N % 64 == 0, K % 32 == 0 (true for all model GEMMs).
// ---------------------------------------------------------------------------
template<int OUTMODE, int ACT>
__global__ __launch_bounds__(128)
void gemm_wmma_kernel(const float* __restrict__ A, const _Float16* __restrict__ Bt,
                      const float* __restrict__ bias, const float* __restrict__ res,
                      float* __restrict__ Cf, _Float16* __restrict__ Ch,
                      int M, int N, int K,
                      long long sA, long long sB, long long sC, float scale)
{
    const int wave  = threadIdx.x >> 5;
    const int lane  = threadIdx.x & 31;
    const int row   = lane & 15;     // M index (A) / N index (B) within a 16-tile
    const int hi    = lane >> 4;     // K-half selector per ISA 16-bit layout
    const int tileM = blockIdx.y * 128 + wave * 32;
    const int tileN = blockIdx.x * 64;
    const long long zb = blockIdx.z;

    const float* ap[2];
    const _Float16* bp[4];
#pragma unroll
    for (int i = 0; i < 2; ++i)
        ap[i] = A + zb * sA + (long long)(tileM + i * 16 + row) * K + hi * 8;
#pragma unroll
    for (int j = 0; j < 4; ++j)
        bp[j] = Bt + zb * sB + (long long)(tileN + j * 16 + row) * K + hi * 8;

    v8f acc[2][4];
#pragma unroll
    for (int i = 0; i < 2; ++i)
#pragma unroll
        for (int j = 0; j < 4; ++j)
            acc[i][j] = (v8f){0.f,0.f,0.f,0.f,0.f,0.f,0.f,0.f};

    for (int k0 = 0; k0 < K; k0 += 32) {
        v16h af[2], bf[4];
#pragma unroll
        for (int i = 0; i < 2; ++i) {
            v8f lo = *(const v8f*)(ap[i] + k0);
            v8f hiv = *(const v8f*)(ap[i] + k0 + 16);
#pragma unroll
            for (int j = 0; j < 8; ++j) {
                af[i][j]     = (_Float16)lo[j];
                af[i][j + 8] = (_Float16)hiv[j];
            }
        }
#pragma unroll
        for (int j = 0; j < 4; ++j) {
            v8h lo = *(const v8h*)(bp[j] + k0);
            v8h hiv = *(const v8h*)(bp[j] + k0 + 16);
#pragma unroll
            for (int t = 0; t < 8; ++t) {
                bf[j][t]     = lo[t];
                bf[j][t + 8] = hiv[t];
            }
        }
#pragma unroll
        for (int i = 0; i < 2; ++i)
#pragma unroll
            for (int j = 0; j < 4; ++j)
                acc[i][j] = __builtin_amdgcn_wmma_f32_16x16x32_f16(
                    false, af[i], false, bf[j], (short)0, acc[i][j], false, false);
    }

#pragma unroll
    for (int i = 0; i < 2; ++i) {
#pragma unroll
        for (int j = 0; j < 4; ++j) {
            const int col = tileN + j * 16 + row;
            const float bv = bias ? bias[col] : 0.0f;
#pragma unroll
            for (int r = 0; r < 8; ++r) {
                const int m = tileM + i * 16 + hi * 8 + r;  // C layout: VGPR r -> M = r (+8 hi)
                float v = acc[i][j][r] * scale + bv;
                if (OUTMODE == 0) {
                    if (res) v += res[zb * sC + (long long)m * N + col];
                }
                if (ACT == 1)      v = 0.5f * v * (1.0f + erff(v * 0.70710678118654752f));
                else if (ACT == 2) v = tanhf(v);
                if (OUTMODE == 0)      Cf[zb * sC + (long long)m * N + col] = v;
                else if (OUTMODE == 1) Ch[zb * sC + (long long)m * N + col] = (_Float16)v;
                else                   Ch[zb * sC + (long long)col * M + m] = (_Float16)v;
            }
        }
    }
}

// ---------------------------------------------------------------------------
// Block reductions (blockDim.x == 128)
// ---------------------------------------------------------------------------
__device__ __forceinline__ float block_sum(float v, float* red)
{
    const int tid = threadIdx.x;
    red[tid] = v; __syncthreads();
    for (int o = 64; o > 0; o >>= 1) {
        if (tid < o) red[tid] += red[tid + o];
        __syncthreads();
    }
    float r = red[0]; __syncthreads();
    return r;
}
__device__ __forceinline__ float block_max(float v, float* red)
{
    const int tid = threadIdx.x;
    red[tid] = v; __syncthreads();
    for (int o = 64; o > 0; o >>= 1) {
        if (tid < o) red[tid] = fmaxf(red[tid], red[tid + o]);
        __syncthreads();
    }
    float r = red[0]; __syncthreads();
    return r;
}

// ---------------------------------------------------------------------------
// LayerNorm over H_DIM=512, one row per block of 128 threads
// ---------------------------------------------------------------------------
__global__ __launch_bounds__(128)
void layernorm_kernel(const float* __restrict__ x, const float* __restrict__ g,
                      const float* __restrict__ b, float* __restrict__ y)
{
    __shared__ float red[128];
    const long long r = blockIdx.x;
    const float* xr = x + r * H_DIM;
    float v[4]; float s = 0.f;
#pragma unroll
    for (int i = 0; i < 4; ++i) { v[i] = xr[threadIdx.x + i * 128]; s += v[i]; }
    const float mu = block_sum(s, red) * (1.0f / H_DIM);
    float q = 0.f;
#pragma unroll
    for (int i = 0; i < 4; ++i) { float d = v[i] - mu; q += d * d; }
    const float var = block_sum(q, red) * (1.0f / H_DIM);
    const float rstd = rsqrtf(var + 1e-12f);
#pragma unroll
    for (int i = 0; i < 4; ++i) {
        int c = threadIdx.x + i * 128;
        y[r * H_DIM + c] = (v[i] - mu) * rstd * g[c] + b[c];
    }
}

// ---------------------------------------------------------------------------
// Embedding + (faithful buggy) positional encoding + LayerNorm
// ---------------------------------------------------------------------------
__global__ __launch_bounds__(128)
void embed_ln_kernel(const int* __restrict__ tok, const float* __restrict__ table,
                     const float* __restrict__ g, const float* __restrict__ b,
                     float* __restrict__ y)
{
    __shared__ float red[128];
    const long long r = blockIdx.x;          // b*SEQ + s
    const int s = (int)(r & (SEQ - 1));
    const int t = tok[r];
    const float* tr = table + (long long)t * H_DIM;
    const float p = (float)s;
    const float freq = powf(10000.0f, 2.0f * floorf(p * 0.5f) * (1.0f / H_DIM));
    const float val = p / freq;
    const float sv = sinf(val), cv = cosf(val);
    float v[4]; float sum = 0.f;
#pragma unroll
    for (int i = 0; i < 4; ++i) {
        int c = threadIdx.x + i * 128;
        float e = tr[c] + (((c & 1) == 0) ? sv : cv);
        v[i] = e; sum += e;
    }
    const float mu = block_sum(sum, red) * (1.0f / H_DIM);
    float q = 0.f;
#pragma unroll
    for (int i = 0; i < 4; ++i) { float d = v[i] - mu; q += d * d; }
    const float var = block_sum(q, red) * (1.0f / H_DIM);
    const float rstd = rsqrtf(var + 1e-12f);
#pragma unroll
    for (int i = 0; i < 4; ++i) {
        int c = threadIdx.x + i * 128;
        y[r * H_DIM + c] = (v[i] - mu) * rstd * g[c] + b[c];
    }
}

// ---------------------------------------------------------------------------
// Row softmax over 512 cols; optional causal mask (col > q masked)
// Row index = (b*8 + h)*512 + q
// ---------------------------------------------------------------------------
__global__ __launch_bounds__(128)
void softmax_kernel(float* __restrict__ sc, int masked)
{
    __shared__ float red[128];
    const long long row = blockIdx.x;
    const int q = (int)(row & (SEQ - 1));
    float* p = sc + row * SEQ;
    float v[4]; float mx = -INFINITY;
#pragma unroll
    for (int i = 0; i < 4; ++i) {
        int c = threadIdx.x + i * 128;
        float x = p[c];
        if (masked && c > q) x = -INFINITY;
        v[i] = x; mx = fmaxf(mx, x);
    }
    const float rmax = block_max(mx, red);
    float e[4]; float s = 0.f;
#pragma unroll
    for (int i = 0; i < 4; ++i) { e[i] = expf(v[i] - rmax); s += e[i]; }
    const float inv = 1.0f / block_sum(s, red);
#pragma unroll
    for (int i = 0; i < 4; ++i) p[threadIdx.x + i * 128] = e[i] * inv;
}

// ---------------------------------------------------------------------------
// Faithful contiguous reshape (B,512,512) <-> (B,8,512,64):
//   src row s = h*64 + q/8, col c = (q%8)*64 + d
// ---------------------------------------------------------------------------
__global__ void gather_heads_kernel(const float* __restrict__ x, float* __restrict__ heads)
{
    for (int t = blockIdx.x * blockDim.x + threadIdx.x; t < HTOT; t += gridDim.x * blockDim.x) {
        int d = t & 63, q = (t >> 6) & 511, h = (t >> 15) & 7, b = t >> 18;
        long long src = ((long long)(b * SEQ + h * 64 + (q >> 3))) * H_DIM + (q & 7) * 64 + d;
        heads[t] = x[src];
    }
}
__global__ void scatter_heads_kernel(const float* __restrict__ heads, float* __restrict__ y)
{
    for (int t = blockIdx.x * blockDim.x + threadIdx.x; t < HTOT; t += gridDim.x * blockDim.x) {
        int d = t & 63, q = (t >> 6) & 511, h = (t >> 15) & 7, b = t >> 18;
        long long dst = ((long long)(b * SEQ + h * 64 + (q >> 3))) * H_DIM + (q & 7) * 64 + d;
        y[dst] = heads[t];
    }
}

// ---------------------------------------------------------------------------
// Weight conversion: fp32 [K][N] -> f16 transposed [N][K]
// ---------------------------------------------------------------------------
__global__ void convert_wT_kernel(const float* __restrict__ W, _Float16* __restrict__ Wt,
                                  int K, int N)
{
    long long total = (long long)K * N;
    for (long long i = blockIdx.x * (long long)blockDim.x + threadIdx.x; i < total;
         i += (long long)gridDim.x * blockDim.x) {
        long long n = i / K, k = i % K;
        Wt[i] = (_Float16)W[k * N + n];
    }
}

// ---------------------------------------------------------------------------
// Host-side launchers (block tile = 128x64, 4 waves of 32x64)
// ---------------------------------------------------------------------------
static void run_gemm(int act, const float* A, const _Float16* Bt, const float* bias,
                     const float* res, float* C, int M, int N, int K,
                     long long sA, long long sB, long long sC, int batch, float scale,
                     hipStream_t st)
{
    dim3 g(N / 64, M / 128, batch), blk(128);
    if (act == 1)
        gemm_wmma_kernel<0, 1><<<g, blk, 0, st>>>(A, Bt, bias, res, C, nullptr, M, N, K, sA, sB, sC, scale);
    else if (act == 2)
        gemm_wmma_kernel<0, 2><<<g, blk, 0, st>>>(A, Bt, bias, res, C, nullptr, M, N, K, sA, sB, sC, scale);
    else
        gemm_wmma_kernel<0, 0><<<g, blk, 0, st>>>(A, Bt, bias, res, C, nullptr, M, N, K, sA, sB, sC, scale);
}
static void run_gemm_h(int transposed, const float* A, const _Float16* Bt, const float* bias,
                       _Float16* C, int M, int N, int K,
                       long long sA, long long sB, long long sC, int batch, hipStream_t st)
{
    dim3 g(N / 64, M / 128, batch), blk(128);
    if (transposed)
        gemm_wmma_kernel<2, 0><<<g, blk, 0, st>>>(A, Bt, bias, nullptr, nullptr, C, M, N, K, sA, sB, sC, 1.0f);
    else
        gemm_wmma_kernel<1, 0><<<g, blk, 0, st>>>(A, Bt, bias, nullptr, nullptr, C, M, N, K, sA, sB, sC, 1.0f);
}

extern "C" void kernel_launch(void* const* d_in, const int* in_sizes, int n_in,
                              void* d_out, int out_size, void* d_ws, size_t ws_size,
                              hipStream_t stream)
{
    (void)in_sizes; (void)n_in; (void)out_size; (void)ws_size;
    auto F = [&](int i) { return (const float*)d_in[i]; };
    const int* src_tok = (const int*)d_in[0];
    const int* tgt_tok = (const int*)d_in[1];
    const float* embed_table = F(2);
    const float* emb_g = F(3), *emb_b = F(4);
    // encoder params
    const float* e_ln1g = F(5), *e_ln1b = F(6), *e_ln2g = F(7), *e_ln2b = F(8);
    const float* e_wq = F(9),  *e_bq = F(10), *e_wk = F(11), *e_bk = F(12);
    const float* e_wv = F(13), *e_bv = F(14), *e_wo = F(15), *e_bo = F(16);
    const float* e_w1 = F(17), *e_b1 = F(18), *e_w2 = F(19), *e_b2 = F(20);
    // decoder params
    const float* d_ln1g = F(21), *d_ln1b = F(22), *d_ln2g = F(23), *d_ln2b = F(24);
    const float* d_ln3g = F(25), *d_ln3b = F(26);
    const float* s_wq = F(27), *s_bq = F(28), *s_wk = F(29), *s_bk = F(30);
    const float* s_wv = F(31), *s_bv = F(32), *s_wo = F(33), *s_bo = F(34);
    const float* c_wq = F(35), *c_bq = F(36), *c_wk = F(37), *c_bk = F(38);
    const float* c_wv = F(39), *c_bv = F(40), *c_wo = F(41), *c_bo = F(42);
    const float* d_w1 = F(43), *d_b1 = F(44), *d_w2 = F(45), *d_b2 = F(46);
    const float* gen_w = F(47), *gen_b = F(48);

    // -------------------- workspace layout --------------------
    char* base = (char*)d_ws;
    size_t off = 0;
    auto alloc = [&](size_t bytes) -> void* {
        void* p = base + off;
        off += (bytes + 255) & ~(size_t)255;
        return p;
    };
    _Float16* e_wqT = (_Float16*)alloc(64 * 64 * 2);
    _Float16* e_wkT = (_Float16*)alloc(64 * 64 * 2);
    _Float16* e_wvT = (_Float16*)alloc(64 * 64 * 2);
    _Float16* e_woT = (_Float16*)alloc(512 * 512 * 2);
    _Float16* e_w1T = (_Float16*)alloc((size_t)512 * 2048 * 2);
    _Float16* e_w2T = (_Float16*)alloc((size_t)2048 * 512 * 2);
    _Float16* s_wqT = (_Float16*)alloc(64 * 64 * 2);
    _Float16* s_wkT = (_Float16*)alloc(64 * 64 * 2);
    _Float16* s_wvT = (_Float16*)alloc(64 * 64 * 2);
    _Float16* s_woT = (_Float16*)alloc(512 * 512 * 2);
    _Float16* c_wqT = (_Float16*)alloc(64 * 64 * 2);
    _Float16* c_wkT = (_Float16*)alloc(64 * 64 * 2);
    _Float16* c_wvT = (_Float16*)alloc(64 * 64 * 2);
    _Float16* c_woT = (_Float16*)alloc(512 * 512 * 2);
    _Float16* d_w1T = (_Float16*)alloc((size_t)512 * 2048 * 2);
    _Float16* d_w2T = (_Float16*)alloc((size_t)2048 * 512 * 2);
    _Float16* genT  = (_Float16*)alloc((size_t)VOCABN * 512 * 2);

    float* xe_a   = (float*)alloc((size_t)ROWS * H_DIM * 4);
    float* xe_b   = (float*)alloc((size_t)ROWS * H_DIM * 4);
    float* xd_a   = (float*)alloc((size_t)ROWS * H_DIM * 4);
    float* xd_b   = (float*)alloc((size_t)ROWS * H_DIM * 4);
    float* tmpN   = (float*)alloc((size_t)ROWS * H_DIM * 4);
    float* headsIn  = (float*)alloc((size_t)HTOT * 4);
    float* headsQ   = (float*)alloc((size_t)HTOT * 4);
    float* concatB  = (float*)alloc((size_t)ROWS * H_DIM * 4);
    float* encHeads = (float*)alloc((size_t)HTOT * 4);
    float* scores = (float*)alloc((size_t)BATCHN * NHEADS * SEQ * SEQ * 4);
    float* ffnb   = (float*)alloc((size_t)ROWS * DFF * 4);
    _Float16* k_h  = (_Float16*)alloc((size_t)HTOT * 2);
    _Float16* vT_h = (_Float16*)alloc((size_t)HTOT * 2);
    _Float16* ck_h  = (_Float16*)alloc((size_t)HTOT * 2);
    _Float16* cvT_h = (_Float16*)alloc((size_t)HTOT * 2);

    // -------------------- weight conversion (f16, [N][K]) --------------------
    auto conv = [&](const float* W, _Float16* Wt, int K, int N) {
        long long total = (long long)K * N;
        int blocks = (int)((total + 255) / 256);
        if (blocks > 65535) blocks = 65535;
        convert_wT_kernel<<<blocks, 256, 0, stream>>>(W, Wt, K, N);
    };
    conv(e_wq, e_wqT, 64, 64);   conv(e_wk, e_wkT, 64, 64);   conv(e_wv, e_wvT, 64, 64);
    conv(e_wo, e_woT, 512, 512); conv(e_w1, e_w1T, 512, 2048); conv(e_w2, e_w2T, 2048, 512);
    conv(s_wq, s_wqT, 64, 64);   conv(s_wk, s_wkT, 64, 64);   conv(s_wv, s_wvT, 64, 64);
    conv(s_wo, s_woT, 512, 512);
    conv(c_wq, c_wqT, 64, 64);   conv(c_wk, c_wkT, 64, 64);   conv(c_wv, c_wvT, 64, 64);
    conv(c_wo, c_woT, 512, 512);
    conv(d_w1, d_w1T, 512, 2048); conv(d_w2, d_w2T, 2048, 512);
    conv(gen_w, genT, 512, VOCABN);

    const long long HS = (long long)SEQ * DPH;       // per-(b,h) qkv stride
    const long long SS = (long long)SEQ * SEQ;       // per-(b,h) score stride
    const int BH = BATCHN * NHEADS;                  // 64
    const float iscale = 0.125f;                     // 1/sqrt(64)

    // -------------------- encoder --------------------
    float* x = xe_a; float* xalt = xe_b;
    embed_ln_kernel<<<ROWS, 128, 0, stream>>>(src_tok, embed_table, emb_g, emb_b, x);
    for (int l = 0; l < 6; ++l) {
        layernorm_kernel<<<ROWS, 128, 0, stream>>>(x, e_ln1g, e_ln1b, tmpN);
        gather_heads_kernel<<<8192, 256, 0, stream>>>(tmpN, headsIn);
        run_gemm(0, headsIn, e_wqT, e_bq, nullptr, headsQ, SEQ, DPH, DPH, HS, 0, HS, BH, 1.0f, stream);
        run_gemm_h(0, headsIn, e_wkT, e_bk, k_h,  SEQ, DPH, DPH, HS, 0, HS, BH, stream);
        run_gemm_h(1, headsIn, e_wvT, e_bv, vT_h, SEQ, DPH, DPH, HS, 0, HS, BH, stream);
        run_gemm(0, headsQ, k_h, nullptr, nullptr, scores, SEQ, SEQ, DPH, HS, HS, SS, BH, iscale, stream);
        softmax_kernel<<<BH * SEQ, 128, 0, stream>>>(scores, 0);
        run_gemm(0, scores, vT_h, nullptr, nullptr, headsIn, SEQ, DPH, SEQ, SS, HS, HS, BH, 1.0f, stream);
        scatter_heads_kernel<<<8192, 256, 0, stream>>>(headsIn, concatB);
        run_gemm(0, concatB, e_woT, e_bo, x, xalt, ROWS, H_DIM, H_DIM, 0, 0,
                 (long long)ROWS * H_DIM, 1, 1.0f, stream);
        { float* t = x; x = xalt; xalt = t; }
        layernorm_kernel<<<ROWS, 128, 0, stream>>>(x, e_ln2g, e_ln2b, tmpN);
        run_gemm(1, tmpN, e_w1T, e_b1, nullptr, ffnb, ROWS, DFF, H_DIM, 0, 0,
                 (long long)ROWS * DFF, 1, 1.0f, stream);
        run_gemm(0, ffnb, e_w2T, e_b2, x, xalt, ROWS, H_DIM, DFF, 0, 0,
                 (long long)ROWS * H_DIM, 1, 1.0f, stream);
        { float* t = x; x = xalt; xalt = t; }
    }
    float* enc_out = x;

    // -------------------- decoder --------------------
    float* xd = xd_a; float* xdalt = xd_b;
    embed_ln_kernel<<<ROWS, 128, 0, stream>>>(tgt_tok, embed_table, emb_g, emb_b, xd);
    // cross-attention K/V depend only on encoder output + shared weights: hoist
    gather_heads_kernel<<<8192, 256, 0, stream>>>(enc_out, encHeads);
    run_gemm_h(0, encHeads, c_wkT, c_bk, ck_h,  SEQ, DPH, DPH, HS, 0, HS, BH, stream);
    run_gemm_h(1, encHeads, c_wvT, c_bv, cvT_h, SEQ, DPH, DPH, HS, 0, HS, BH, stream);

    for (int l = 0; l < 6; ++l) {
        // masked self-attention
        layernorm_kernel<<<ROWS, 128, 0, stream>>>(xd, d_ln1g, d_ln1b, tmpN);
        gather_heads_kernel<<<8192, 256, 0, stream>>>(tmpN, headsIn);
        run_gemm(0, headsIn, s_wqT, s_bq, nullptr, headsQ, SEQ, DPH, DPH, HS, 0, HS, BH, 1.0f, stream);
        run_gemm_h(0, headsIn, s_wkT, s_bk, k_h,  SEQ, DPH, DPH, HS, 0, HS, BH, stream);
        run_gemm_h(1, headsIn, s_wvT, s_bv, vT_h, SEQ, DPH, DPH, HS, 0, HS, BH, stream);
        run_gemm(0, headsQ, k_h, nullptr, nullptr, scores, SEQ, SEQ, DPH, HS, HS, SS, BH, iscale, stream);
        softmax_kernel<<<BH * SEQ, 128, 0, stream>>>(scores, 1);
        run_gemm(0, scores, vT_h, nullptr, nullptr, headsIn, SEQ, DPH, SEQ, SS, HS, HS, BH, 1.0f, stream);
        scatter_heads_kernel<<<8192, 256, 0, stream>>>(headsIn, concatB);
        run_gemm(0, concatB, s_woT, s_bo, xd, xdalt, ROWS, H_DIM, H_DIM, 0, 0,
                 (long long)ROWS * H_DIM, 1, 1.0f, stream);
        { float* t = xd; xd = xdalt; xdalt = t; }
        // cross-attention
        layernorm_kernel<<<ROWS, 128, 0, stream>>>(xd, d_ln2g, d_ln2b, tmpN);
        gather_heads_kernel<<<8192, 256, 0, stream>>>(tmpN, headsIn);
        run_gemm(0, headsIn, c_wqT, c_bq, nullptr, headsQ, SEQ, DPH, DPH, HS, 0, HS, BH, 1.0f, stream);
        run_gemm(0, headsQ, ck_h, nullptr, nullptr, scores, SEQ, SEQ, DPH, HS, HS, SS, BH, iscale, stream);
        softmax_kernel<<<BH * SEQ, 128, 0, stream>>>(scores, 0);
        run_gemm(0, scores, cvT_h, nullptr, nullptr, headsIn, SEQ, DPH, SEQ, SS, HS, HS, BH, 1.0f, stream);
        scatter_heads_kernel<<<8192, 256, 0, stream>>>(headsIn, concatB);
        run_gemm(0, concatB, c_woT, c_bo, xd, xdalt, ROWS, H_DIM, H_DIM, 0, 0,
                 (long long)ROWS * H_DIM, 1, 1.0f, stream);
        { float* t = xd; xd = xdalt; xdalt = t; }
        // FFN
        layernorm_kernel<<<ROWS, 128, 0, stream>>>(xd, d_ln3g, d_ln3b, tmpN);
        run_gemm(1, tmpN, d_w1T, d_b1, nullptr, ffnb, ROWS, DFF, H_DIM, 0, 0,
                 (long long)ROWS * DFF, 1, 1.0f, stream);
        run_gemm(0, ffnb, d_w2T, d_b2, xd, xdalt, ROWS, H_DIM, DFF, 0, 0,
                 (long long)ROWS * H_DIM, 1, 1.0f, stream);
        { float* t = xd; xd = xdalt; xdalt = t; }
    }

    // -------------------- generator: tanh(q @ gen_w + gen_b) --------------------
    run_gemm(2, xd, genT, gen_b, nullptr, (float*)d_out, ROWS, VOCABN, H_DIM,
             0, 0, (long long)ROWS * VOCABN, 1, 1.0f, stream);
}
